// VisibleNNWithSkip_57621281243253
// MI455X (gfx1250) — matrix operand order repaired and checked by reading, not compiled, not run on hardware
//
#include <hip/hip_runtime.h>
#include <math.h>

typedef __bf16 bf16_t;
typedef __attribute__((ext_vector_type(16))) __bf16 v16bf;
typedef __attribute__((ext_vector_type(4)))  __bf16 bf16x4;
typedef __attribute__((ext_vector_type(8)))  float  v8f;
typedef __attribute__((ext_vector_type(4)))  float  float4v;

#define TILE_M 128
#define TILE_N 64
#define TILE_K 32

// ---- staging helpers: 256 threads stage A(128x32) and B(64x32) tiles ----
// per thread: A = 4 float4 (rows aRow0 + 32*i, cols c4..c4+3), B = 2 float4 of W and M.
__device__ __forceinline__
void stage_loads(const float* __restrict__ A, const float* __restrict__ W,
                 const float* __restrict__ Mk, int F,
                 int aRow0, int bRow0, int k0, int c4,
                 float4v ar[4], float4v wr[2], float4v mr[2])
{
    #pragma unroll
    for (int i = 0; i < 4; ++i)
        ar[i] = *(const float4v*)(A + (size_t)(aRow0 + 32 * i) * F + (k0 + c4));
    #pragma unroll
    for (int i = 0; i < 2; ++i) {
        size_t gi = (size_t)(bRow0 + 32 * i) * F + (k0 + c4);
        wr[i] = *(const float4v*)(W + gi);
        mr[i] = *(const float4v*)(Mk + gi);
    }
}

template<int RELU>
__device__ __forceinline__
void stage_store(bf16_t* __restrict__ la, bf16_t* __restrict__ lb, int tid,
                 const float4v ar[4], const float4v wr[2], const float4v mr[2])
{
    #pragma unroll
    for (int i = 0; i < 4; ++i) {
        bf16x4 h;
        #pragma unroll
        for (int e = 0; e < 4; ++e) {
            float v = ar[i][e];
            if (RELU) v = fmaxf(v, 0.0f);
            h[e] = (bf16_t)v;
        }
        *(bf16x4*)(la + (size_t)(tid + i * 256) * 4) = h;   // 8B aligned, contiguous
    }
    #pragma unroll
    for (int i = 0; i < 2; ++i) {
        bf16x4 h;
        #pragma unroll
        for (int e = 0; e < 4; ++e)
            h[e] = (bf16_t)(wr[i][e] * mr[i][e]);
        *(bf16x4*)(lb + (size_t)(tid + i * 256) * 4) = h;
    }
}

// C[row, col] = sum_f act(A[row,f]) * (W[col,f]*M[col,f]) + bias[col]
// A: (Brows x F) f32 row-major; W,M: (O x F) f32 row-major; C: pre-activation f32.
// Double-buffered LDS pipeline: next-tile global loads are issued before this
// tile's WMMAs so the s_wait_loadcnt lands after the matrix ops.
template<int RELU>
__global__ __launch_bounds__(256)
void gemm_bf16_wmma(const float* __restrict__ A,
                    const float* __restrict__ W,
                    const float* __restrict__ Mk,
                    const float* __restrict__ bias,
                    float* __restrict__ C,
                    int F, int O)
{
    __shared__ bf16_t la[2][TILE_M * TILE_K];   // [m][k] row-major, bf16
    __shared__ bf16_t lb[2][TILE_N * TILE_K];   // [n][k] row-major, (W.*M) bf16

    const int tid  = threadIdx.x;
    const int lane = tid & 31;
    const int wave = tid >> 5;
    const int wm   = wave & 3;    // 4 waves over M (4*32 = 128)
    const int wn   = wave >> 2;   // 2 waves over N (2*32 = 64)
    const int ln16 = lane & 15;
    const int hi   = lane >> 4;

    const int rowBase = blockIdx.y * TILE_M;
    const int colBase = blockIdx.x * TILE_N;

    const int aRow0 = rowBase + (tid >> 3);
    const int bRow0 = colBase + (tid >> 3);
    const int c4    = (tid & 7) * 4;

    v8f acc[2][2] = {};
    float4v ar[4], wr[2], mr[2];

    // prologue: stage tile 0 into buffer 0
    stage_loads(A, W, Mk, F, aRow0, bRow0, 0, c4, ar, wr, mr);
    stage_store<RELU>(la[0], lb[0], tid, ar, wr, mr);
    __syncthreads();

    const int nk = F / TILE_K;
    for (int kt = 0; kt < nk; ++kt) {
        const int cur = kt & 1;
        const bool hasNext = (kt + 1) < nk;

        // issue next tile's global loads (latency hidden behind WMMAs below)
        if (hasNext)
            stage_loads(A, W, Mk, F, aRow0, bRow0, (kt + 1) * TILE_K, c4, ar, wr, mr);

        // ---- fragments per CDNA5 WMMA 16-bit layouts (ISA 7.12.2) ----
        v16bf af[2], bfr[2];
        #pragma unroll
        for (int i = 0; i < 2; ++i) {
            // A 16x32: lane m = ln16; element e -> K = (e%8) + (e/8)*16 + hi*8
            const bf16_t* p = &la[cur][(wm * 32 + i * 16 + ln16) * TILE_K];
            #pragma unroll
            for (int g = 0; g < 2; ++g)
                #pragma unroll
                for (int e = 0; e < 8; ++e)
                    af[i][g * 8 + e] = p[g * 16 + hi * 8 + e];   // 2 contiguous 16B runs
        }
        #pragma unroll
        for (int j = 0; j < 2; ++j) {
            // B 32x16: lane n = ln16; element e -> K = (e%8) + (e/8)*16 + hi*8
            const bf16_t* p = &lb[cur][(wn * 32 + j * 16 + ln16) * TILE_K];
            #pragma unroll
            for (int g = 0; g < 2; ++g)
                #pragma unroll
                for (int e = 0; e < 8; ++e)
                    bfr[j][g * 8 + e] = p[g * 16 + hi * 8 + e];
        }

        #pragma unroll
        for (int i = 0; i < 2; ++i)
            #pragma unroll
            for (int j = 0; j < 2; ++j)
                acc[i][j] = __builtin_amdgcn_wmma_f32_16x16x32_bf16(
                    false, af[i], false, bfr[j], (short)0, acc[i][j], false, false);

        // convert + park next tile in the other LDS buffer
        if (hasNext)
            stage_store<RELU>(la[cur ^ 1], lb[cur ^ 1], tid, ar, wr, mr);
        __syncthreads();
    }

    // ---- epilogue: C/D layout -> row = i*16 + hi*8 + r, col = j*16 + ln16 ----
    #pragma unroll
    for (int j = 0; j < 2; ++j) {
        int col = colBase + wn * 32 + j * 16 + ln16;
        float bv = bias[col];
        #pragma unroll
        for (int i = 0; i < 2; ++i) {
            #pragma unroll
            for (int r = 0; r < 8; ++r) {
                int row = rowBase + wm * 32 + i * 16 + hi * 8 + r;
                C[(size_t)row * O + col] = acc[i][j][r] + bv;
            }
        }
    }
}

// Layer 5: act5[b] = sum_{f<256} relu(act4[b,f]) * w5[f] * m5[f] + b5[0].
__global__ __launch_bounds__(256)
void layer5_kernel(const float* __restrict__ act4,
                   const float* __restrict__ w5, const float* __restrict__ m5,
                   const float* __restrict__ b5, float* __restrict__ act5)
{
    int wave = threadIdx.x >> 5;
    int lane = threadIdx.x & 31;
    int row  = blockIdx.x * 8 + wave;
    float s = 0.0f;
    #pragma unroll
    for (int i = 0; i < 8; ++i) {
        int f = lane + i * 32;
        s += fmaxf(act4[(size_t)row * 256 + f], 0.0f) * w5[f] * m5[f];
    }
    #pragma unroll
    for (int m = 16; m > 0; m >>= 1) s += __shfl_xor(s, m, 32);
    if (lane == 0) act5[row] = s + b5[0];
}

// Skip projection over concatenated pre-acts (7680) + gated combine with act5.
__global__ __launch_bounds__(256)
void skip_final_kernel(const float* __restrict__ p0, const float* __restrict__ p1,
                       const float* __restrict__ p2, const float* __restrict__ p3,
                       const float* __restrict__ wsk, const float* __restrict__ msk,
                       const float* __restrict__ bsk, const float* __restrict__ act5,
                       const float* __restrict__ alpha, float* __restrict__ out)
{
    int b = blockIdx.x;
    float s = 0.0f;
    for (int j = threadIdx.x; j < 7680; j += 256) {
        float a;
        if      (j < 4096) a = p0[(size_t)b * 4096 + j];
        else if (j < 6144) a = p1[(size_t)b * 2048 + (j - 4096)];
        else if (j < 7168) a = p2[(size_t)b * 1024 + (j - 6144)];
        else               a = p3[(size_t)b *  512 + (j - 7168)];
        s += a * wsk[j] * msk[j];
    }
    __shared__ float red[256];
    red[threadIdx.x] = s;
    __syncthreads();
    #pragma unroll
    for (int off = 128; off > 0; off >>= 1) {
        if (threadIdx.x < off) red[threadIdx.x] += red[threadIdx.x + off];
        __syncthreads();
    }
    if (threadIdx.x == 0) {
        float gate = 1.0f / (1.0f + __expf(-alpha[0]));
        out[b] = act5[b] + gate * (red[0] + bsk[0]);
    }
}

extern "C" void kernel_launch(void* const* d_in, const int* in_sizes, int n_in,
                              void* d_out, int out_size, void* d_ws, size_t ws_size,
                              hipStream_t stream) {
    (void)in_sizes; (void)n_in; (void)out_size; (void)ws_size;
    // setup_inputs() dict order: x, (w,b,m)x6, m_skip, w_skip, b_skip, alpha
    const float* x = (const float*)d_in[0];
    const float *w[6], *b[6], *m[6];
    for (int i = 0; i < 6; ++i) {
        w[i] = (const float*)d_in[1 + 3 * i];
        b[i] = (const float*)d_in[2 + 3 * i];
        m[i] = (const float*)d_in[3 + 3 * i];
    }
    const float* m_skip = (const float*)d_in[19];
    const float* w_skip = (const float*)d_in[20];
    const float* b_skip = (const float*)d_in[21];
    const float* alpha  = (const float*)d_in[22];

    const int B = 4096;
    float* act0 = (float*)d_ws;                 // B x 4096 pre-act
    float* act1 = act0 + (size_t)B * 4096;      // B x 2048
    float* act2 = act1 + (size_t)B * 2048;      // B x 1024
    float* act3 = act2 + (size_t)B * 1024;      // B x 512
    float* act4 = act3 + (size_t)B * 512;       // B x 256
    float* act5 = act4 + (size_t)B * 256;       // B x 1 (logit base)

    dim3 blk(256);
    dim3 gm(0, B / TILE_M);

    gm.x = 4096 / TILE_N; gemm_bf16_wmma<0><<<gm, blk, 0, stream>>>(x,    w[0], m[0], b[0], act0, 8192, 4096);
    gm.x = 2048 / TILE_N; gemm_bf16_wmma<1><<<gm, blk, 0, stream>>>(act0, w[1], m[1], b[1], act1, 4096, 2048);
    gm.x = 1024 / TILE_N; gemm_bf16_wmma<1><<<gm, blk, 0, stream>>>(act1, w[2], m[2], b[2], act2, 2048, 1024);
    gm.x =  512 / TILE_N; gemm_bf16_wmma<1><<<gm, blk, 0, stream>>>(act2, w[3], m[3], b[3], act3, 1024,  512);
    gm.x =  256 / TILE_N; gemm_bf16_wmma<1><<<gm, blk, 0, stream>>>(act3, w[4], m[4], b[4], act4,  512,  256);

    layer5_kernel<<<dim3(B / 8), blk, 0, stream>>>(act4, w[5], m[5], b[5], act5);
    skip_final_kernel<<<dim3(B), blk, 0, stream>>>(act0, act1, act2, act3,
                                                   w_skip, m_skip, b_skip, act5, alpha,
                                                   (float*)d_out);
}
